// SimilarityGraphBlock_Fast_49177375539802
// MI455X (gfx1250) — matrix operand order repaired and checked by reading, not compiled, not run on hardware
//
#include <hip/hip_runtime.h>
#include <hip/hip_bf16.h>
#include <math.h>

// ---------------------------------------------------------------------------
// Fused: out = LayerNorm( x + W2^T GELU(W1^T x + b1) + b2 ) * gamma + beta
// N = 262144 rows, DIM = 256.  f16 WMMA (16x16x32) with fp32 accumulation.
// 64-row tile per block (8 waves): wave = 2 row-tiles x 4 col-tiles,
// 128 WMMAs/wave.  LayerNorm done register-resident with shfl reductions.
// ---------------------------------------------------------------------------

typedef __attribute__((ext_vector_type(16))) _Float16 v16h;
typedef __attribute__((ext_vector_type(8)))  _Float16 v8h;
typedef __attribute__((ext_vector_type(4)))  _Float16 v4h;
typedef __attribute__((ext_vector_type(8)))  float    v8f;
typedef __attribute__((ext_vector_type(4)))  float    v4f;

#define DIM    256
#define ROWS   64           // rows per workgroup tile
#define LDH    (DIM + 8)    // f16 LDS row stride (halves): 528B rows, 16B aligned
#define KSTEPS 8            // 256 / 32

// Flat offset (in halves) of the WMMA B-fragment for (col-tile ct, k-step ks, lane).
__device__ __forceinline__ size_t bfragOff(int ct, int ks, int lane) {
    return ((size_t)((ct * KSTEPS + ks) * 32 + lane)) * 16;
}

// ---------------------------------------------------------------------------
// Pre-swizzle 256x256 fp32 weight (row-major; B[k][n] = W[k*256+n]) into
// per-lane f16 B-fragment layout: 32x16 tile, lanes 0-15 -> N=0..15 with
// K=0..7,16..23 (element j), lanes 16-31 -> same N with K=8..15,24..31.
// ---------------------------------------------------------------------------
__global__ void swizzle_w_kernel(const float* __restrict__ W, _Float16* __restrict__ Wp) {
    int idx  = blockIdx.x * 256 + threadIdx.x;   // 65536 halves
    int j    = idx & 15;
    int lane = (idx >> 4) & 31;
    int ks   = (idx >> 9) & 7;
    int ct   = (idx >> 12) & 15;
    int k = ks * 32 + ((j >> 3) << 4) + ((lane >> 4) << 3) + (j & 7);
    int n = ct * 16 + (lane & 15);
    Wp[idx] = (_Float16)W[k * DIM + n];
}

// 16x32 f16 A-fragment from row-major f16 LDS (stride LDH), per ISA layout:
// lane&15 = M; VGPR0-3 hold K=klo..klo+7, VGPR4-7 hold K=klo+16..klo+23,
// klo = ks*32 + (lane>>4)*8  ->  two ds_load_b128 per lane.
__device__ __forceinline__ v16h loadAfrag(const _Float16* sBuf, int mBase, int ks, int lane) {
    int m   = mBase + (lane & 15);
    int klo = ks * 32 + ((lane >> 4) << 3);
    const _Float16* p = sBuf + m * LDH + klo;
    v8h lo = *(const v8h*)(p);
    v8h hi = *(const v8h*)(p + 16);
    v16h a;
#pragma unroll
    for (int q = 0; q < 8; ++q) { a[q] = lo[q]; a[8 + q] = hi[q]; }
    return a;
}

__global__ __launch_bounds__(256)
void fused_mlp_ln_kernel(const float* __restrict__ x,
                         const _Float16* __restrict__ W1p,
                         const float* __restrict__ b1,
                         const _Float16* __restrict__ W2p,
                         const float* __restrict__ b2,
                         const float* __restrict__ gamma,
                         const float* __restrict__ beta,
                         float* __restrict__ out)
{
    __shared__ _Float16 sX[ROWS * LDH];        // f16 x tile (GEMM1 A operand)
    __shared__ _Float16 sH[ROWS * LDH];        // f16 GELU(h) tile (GEMM2 A operand)
    __shared__ float    sPart[ROWS][4][2];     // per-row per-colgroup partial sum/sumsq
    __shared__ float    sMu[ROWS], sRs[ROWS];  // LN row stats

    const int t    = threadIdx.x;
    const int lane = t & 31;
    const int wave = t >> 5;
    const int rp   = wave & 1;        // row-tile pair: rows [rp*32, rp*32+32)
    const int cg   = wave >> 1;       // column group (0..3): cols [cg*64, cg*64+64)
    const int nLane = lane & 15;
    const int half8 = (lane >> 4) << 3;       // 0 or 8: M offset of upper half-wave
    const long rowBase = (long)blockIdx.x * ROWS;
    const float* xBlk = x + rowBase * DIM;

    // ---- Prologue: coalesced b128 loads of 64x256 fp32 x, convert to f16 LDS.
#pragma unroll
    for (int i = 0; i < 16; ++i) {
        int flat = t + i * 256;                 // float4 index, 4096 total
        int r = flat >> 6;
        int c = (flat & 63) << 2;
        v4f v = __builtin_nontemporal_load((const v4f*)(xBlk + r * DIM + c));
        v4h h; h.x = (_Float16)v.x; h.y = (_Float16)v.y;
               h.z = (_Float16)v.z; h.w = (_Float16)v.w;
        *(v4h*)(&sX[r * LDH + c]) = h;
    }
    __syncthreads();

    const int mB0 = rp * 32;          // row-tile 0 of this wave
    const int mB1 = rp * 32 + 16;     // row-tile 1
    float b1v[4], b2v[4];
#pragma unroll
    for (int j = 0; j < 4; ++j) {
        int col = cg * 64 + j * 16 + nLane;
        b1v[j] = b1[col];
        b2v[j] = b2[col];
    }

    // ---- GEMM1: acc = x @ W1   (wave: 32 rows x 64 cols, 64 WMMAs)
    v8f acc[2][4];
#pragma unroll
    for (int r = 0; r < 2; ++r)
#pragma unroll
        for (int j = 0; j < 4; ++j) acc[r][j] = (v8f){0,0,0,0,0,0,0,0};

#pragma unroll
    for (int ks = 0; ks < KSTEPS; ++ks) {
        if (ks + 1 < KSTEPS)
            __builtin_prefetch(W1p + bfragOff(cg * 4, ks + 1, lane), 0, 3);
        v16h a0 = loadAfrag(sX, mB0, ks, lane);
        v16h a1 = loadAfrag(sX, mB1, ks, lane);
#pragma unroll
        for (int j = 0; j < 4; ++j) {
            v16h b = *(const v16h*)(W1p + bfragOff(cg * 4 + j, ks, lane));
            acc[0][j] = __builtin_amdgcn_wmma_f32_16x16x32_f16(
                false, a0, false, b, (short)0, acc[0][j], false, false);
            acc[1][j] = __builtin_amdgcn_wmma_f32_16x16x32_f16(
                false, a1, false, b, (short)0, acc[1][j], false, false);
        }
    }

    // ---- bias + exact GELU -> f16 h tile in LDS.
    // D layout: VGPR v: lanes 0-15 -> M=v, lanes 16-31 -> M=v+8; N = lane&15.
#pragma unroll
    for (int r = 0; r < 2; ++r) {
        int rBase = rp * 32 + r * 16 + half8;
#pragma unroll
        for (int j = 0; j < 4; ++j) {
            int col = cg * 64 + j * 16 + nLane;
#pragma unroll
            for (int v = 0; v < 8; ++v) {
                float val = acc[r][j][v] + b1v[j];
                float g = 0.5f * val * (1.0f + erff(val * 0.70710678118654752f));
                sH[(rBase + v) * LDH + col] = (_Float16)g;
            }
        }
    }
    __syncthreads();

    // ---- GEMM2: acc2 = h @ W2
    v8f acc2[2][4];
#pragma unroll
    for (int r = 0; r < 2; ++r)
#pragma unroll
        for (int j = 0; j < 4; ++j) acc2[r][j] = (v8f){0,0,0,0,0,0,0,0};

#pragma unroll
    for (int ks = 0; ks < KSTEPS; ++ks) {
        if (ks + 1 < KSTEPS)
            __builtin_prefetch(W2p + bfragOff(cg * 4, ks + 1, lane), 0, 3);
        v16h a0 = loadAfrag(sH, mB0, ks, lane);
        v16h a1 = loadAfrag(sH, mB1, ks, lane);
#pragma unroll
        for (int j = 0; j < 4; ++j) {
            v16h b = *(const v16h*)(W2p + bfragOff(cg * 4 + j, ks, lane));
            acc2[0][j] = __builtin_amdgcn_wmma_f32_16x16x32_f16(
                false, a0, false, b, (short)0, acc2[0][j], false, false);
            acc2[1][j] = __builtin_amdgcn_wmma_f32_16x16x32_f16(
                false, a1, false, b, (short)0, acc2[1][j], false, false);
        }
    }

    // ---- Residual (fp32 x re-read, L2-hot) + b2, in registers.
    //      Row partial sums reduced with shfl across the 16-lane half-wave.
#pragma unroll
    for (int r = 0; r < 2; ++r) {
        int rBase = rp * 32 + r * 16 + half8;
#pragma unroll
        for (int v = 0; v < 8; ++v) {
            float s = 0.f, s2 = 0.f;
#pragma unroll
            for (int j = 0; j < 4; ++j) {
                int col = cg * 64 + j * 16 + nLane;
                float o = acc2[r][j][v] + b2v[j] + xBlk[(rBase + v) * DIM + col];
                acc2[r][j][v] = o;
                s += o;
                s2 += o * o;
            }
#pragma unroll
            for (int m = 1; m < 16; m <<= 1) {
                s  += __shfl_xor(s,  m, 32);
                s2 += __shfl_xor(s2, m, 32);
            }
            if (nLane == 0) {               // lane 0 & lane 16 each own 8 rows
                sPart[rBase + v][cg][0] = s;
                sPart[rBase + v][cg][1] = s2;
            }
        }
    }
    __syncthreads();

    // ---- Row stats.
    if (t < ROWS) {
        float s = 0.f, s2 = 0.f;
#pragma unroll
        for (int i = 0; i < 4; ++i) { s += sPart[t][i][0]; s2 += sPart[t][i][1]; }
        float mu  = s * (1.0f / DIM);
        float var = s2 * (1.0f / DIM) - mu * mu;
        sMu[t] = mu;
        sRs[t] = rsqrtf(var + 1e-5f);
    }
    __syncthreads();

    // ---- Normalize + affine from registers, nontemporal stores.
    float gmv[4], btv[4];
#pragma unroll
    for (int j = 0; j < 4; ++j) {
        int col = cg * 64 + j * 16 + nLane;
        gmv[j] = gamma[col];
        btv[j] = beta[col];
    }
    float* oBlk = out + rowBase * DIM;
#pragma unroll
    for (int r = 0; r < 2; ++r) {
        int rBase = rp * 32 + r * 16 + half8;
#pragma unroll
        for (int v = 0; v < 8; ++v) {
            int row = rBase + v;
            float mu = sMu[row], rs = sRs[row];
#pragma unroll
            for (int j = 0; j < 4; ++j) {
                int col = cg * 64 + j * 16 + nLane;
                float val = (acc2[r][j][v] - mu) * rs * gmv[j] + btv[j];
                __builtin_nontemporal_store(val, &oBlk[row * DIM + col]);
            }
        }
    }
}

// ---------------------------------------------------------------------------
extern "C" void kernel_launch(void* const* d_in, const int* in_sizes, int n_in,
                              void* d_out, int out_size, void* d_ws, size_t ws_size,
                              hipStream_t stream) {
    const float* x     = (const float*)d_in[0];
    const float* W1    = (const float*)d_in[1];
    const float* b1    = (const float*)d_in[2];
    const float* W2    = (const float*)d_in[3];
    const float* b2    = (const float*)d_in[4];
    const float* gamma = (const float*)d_in[5];
    const float* beta  = (const float*)d_in[6];
    float* out = (float*)d_out;

    const int N = in_sizes[0] / DIM;             // 262144

    _Float16* W1p = (_Float16*)d_ws;             // 65536 halves = 128 KB
    _Float16* W2p = W1p + (DIM * DIM);           // next 128 KB

    swizzle_w_kernel<<<DIM * DIM / 256, 256, 0, stream>>>(W1, W1p);
    swizzle_w_kernel<<<DIM * DIM / 256, 256, 0, stream>>>(W2, W2p);

    fused_mlp_ln_kernel<<<N / ROWS, 256, 0, stream>>>(
        x, W1p, b1, W2p, b2, gamma, beta, out);
}
